// EfficientSelfAttention_84928683311933
// MI455X (gfx1250) — matrix operand order repaired and checked by reading, not compiled, not run on hardware
//
#include <hip/hip_runtime.h>

typedef __bf16 bf16;
typedef __attribute__((ext_vector_type(16))) __bf16 v16bf;
typedef __attribute__((ext_vector_type(8)))  __bf16 v8bf;
typedef __attribute__((ext_vector_type(4)))  __bf16 v4bf;
typedef __attribute__((ext_vector_type(8)))  float  v8f;
typedef __attribute__((ext_vector_type(4)))  float  f32x4;

// ---------------------------------------------------------------------------
// WMMA fragment loaders (16x16x32 bf16, contiguous k = 16*(lane/16) + s in both A and B)
// ---------------------------------------------------------------------------
__device__ __forceinline__ v16bf load_afrag(const bf16* A, int lda, int lane) {
  const int row = lane & 15;
  const int kh  = (lane >> 4) * 16;
  const bf16* p = A + (size_t)row * lda;
  union { v16bf v; v8bf h[2]; } u;
  u.h[0] = *reinterpret_cast<const v8bf*>(p + kh);
  u.h[1] = *reinterpret_cast<const v8bf*>(p + kh + 8);
  return u.v;
}

__device__ __forceinline__ v16bf load_bfrag(const bf16* BT, int ldbt, int lane) {
  const int col = lane & 15;
  const int kh  = (lane >> 4) * 16;
  const bf16* p = BT + (size_t)col * ldbt + kh;
  union { v16bf v; v8bf h[2]; } u;
  u.h[0] = *reinterpret_cast<const v8bf*>(p);
  u.h[1] = *reinterpret_cast<const v8bf*>(p + 8);
  return u.v;
}

#define WMMA_BF16(a, b, c) \
  __builtin_amdgcn_wmma_f32_16x16x32_bf16(false, (a), false, (b), (short)0, (c), false, false)

// ---------------------------------------------------------------------------
// CDNA5 async global->LDS copy (CDNA5 ISA):
// 16B per lane, tracked with ASYNCcnt, no VGPR round-trip.
// LDS operand: low 32 bits of the generic pointer == wave-relative LDS addr
// (ISA §10.2 aperture mapping).
// ---------------------------------------------------------------------------
__device__ __forceinline__ void async_copy16(const bf16* gsrc, bf16* ldst) {
  const unsigned loff = (unsigned)(unsigned long long)ldst;
  asm volatile("global_load_async_to_lds_b128 %0, %1, off"
               :: "v"(loff), "v"(gsrc) : "memory");
}
__device__ __forceinline__ void wait_async0() {
  asm volatile("s_wait_asynccnt 0" ::: "memory");
}

// ---------------------------------------------------------------------------
// Generic bf16 GEMM: C[M,N] = A[M,K] (row-major) @ BT[N,K]^T
// 256 threads = 8 waves (4 M x 2 N), block tile 64x128, wave tile 16x64.
// Software pipelined: A tile async->LDS ping-pong, B fragments register
// double-buffered (loads for k+32 issued before the k WMMAs).
// EPI 0: bf16 store, EPI 1: f32 store + bias.
// ---------------------------------------------------------------------------
template <int EPI>
__launch_bounds__(256)
__global__ void gemm_bf16_kernel(const bf16* __restrict__ A, const bf16* __restrict__ BT,
                                 void* __restrict__ Cv, const float* __restrict__ bias,
                                 int K, int lda, int ldbt, int ldc) {
  __shared__ bf16 As[2][64][32];
  const int t = threadIdx.x;
  const int lane = t & 31, wid = t >> 5;
  const int wm = wid >> 1, wn = wid & 1;
  const int row0 = blockIdx.x * 64;
  const int col0 = blockIdx.y * 128;
  v8f acc[4] = {};
  const int lr = t >> 2, lch = (t & 3) * 8;
  const bf16* Ap = A + (size_t)(row0 + lr) * lda + lch;
  bf16* l0 = &As[0][lr][lch];
  bf16* l1 = &As[1][lr][lch];
  const bf16* bb = BT + (size_t)(col0 + wn * 64) * ldbt;

  // prologue: stage k=0
  async_copy16(Ap, l0);
  v16bf b0 = load_bfrag(bb + (size_t)0 * 16 * ldbt, ldbt, lane);
  v16bf b1 = load_bfrag(bb + (size_t)1 * 16 * ldbt, ldbt, lane);
  v16bf b2 = load_bfrag(bb + (size_t)2 * 16 * ldbt, ldbt, lane);
  v16bf b3 = load_bfrag(bb + (size_t)3 * 16 * ldbt, ldbt, lane);
  wait_async0();
  __syncthreads();

  for (int k0 = 0; k0 < K; k0 += 64) {
    // even step: consume As[0] + b*, prefetch k0+32 into As[1] + n*
    v16bf a = load_afrag(&As[0][wm * 16][0], 32, lane);
    const int k1 = k0 + 32;
    async_copy16(Ap + k1, l1);
    v16bf n0 = load_bfrag(bb + (size_t)0 * 16 * ldbt + k1, ldbt, lane);
    v16bf n1 = load_bfrag(bb + (size_t)1 * 16 * ldbt + k1, ldbt, lane);
    v16bf n2 = load_bfrag(bb + (size_t)2 * 16 * ldbt + k1, ldbt, lane);
    v16bf n3 = load_bfrag(bb + (size_t)3 * 16 * ldbt + k1, ldbt, lane);
    acc[0] = WMMA_BF16(a, b0, acc[0]);
    acc[1] = WMMA_BF16(a, b1, acc[1]);
    acc[2] = WMMA_BF16(a, b2, acc[2]);
    acc[3] = WMMA_BF16(a, b3, acc[3]);
    wait_async0();
    __syncthreads();

    // odd step: consume As[1] + n*, prefetch k0+64 into As[0] + b*
    a = load_afrag(&As[1][wm * 16][0], 32, lane);
    const int k2 = k0 + 64;
    if (k2 < K) {
      async_copy16(Ap + k2, l0);
      b0 = load_bfrag(bb + (size_t)0 * 16 * ldbt + k2, ldbt, lane);
      b1 = load_bfrag(bb + (size_t)1 * 16 * ldbt + k2, ldbt, lane);
      b2 = load_bfrag(bb + (size_t)2 * 16 * ldbt + k2, ldbt, lane);
      b3 = load_bfrag(bb + (size_t)3 * 16 * ldbt + k2, ldbt, lane);
    }
    acc[0] = WMMA_BF16(a, n0, acc[0]);
    acc[1] = WMMA_BF16(a, n1, acc[1]);
    acc[2] = WMMA_BF16(a, n2, acc[2]);
    acc[3] = WMMA_BF16(a, n3, acc[3]);
    wait_async0();
    __syncthreads();
  }

  const int lo = lane & 15, hi = lane >> 4;
  const int rbase = row0 + wm * 16 + hi * 8;
  const int cb = col0 + wn * 64;
#pragma unroll
  for (int s = 0; s < 4; ++s)
#pragma unroll
    for (int j = 0; j < 8; ++j) {
      const int rr = rbase + j, cc = cb + s * 16 + lo;
      if (EPI == 0)
        ((bf16*)Cv)[(size_t)rr * ldc + cc] = (bf16)acc[s][j];
      else
        ((float*)Cv)[(size_t)rr * ldc + cc] = acc[s][j] + bias[cc];
    }
}

// ---------------------------------------------------------------------------
// Conv-as-patch-GEMM. stride==kernel==8, SAME => zero pad. Rows = B*64 patch
// pixels, K = 8*8*512 ordered (r1,r2,ci). Split-K over gridDim.z into
// deterministic partial buffers. Same async/double-buffered pipeline.
// ---------------------------------------------------------------------------
__device__ __forceinline__ const bf16* conv_src(const bf16* xb, int gb, int pi, int pj,
                                                int k, int lch) {
  const int rr = k >> 9, r1 = rr >> 3, r2 = rr & 7, ci0 = k & 511;
  const int npix = ((pi * 8 + r1) << 6) + pj * 8 + r2;
  return xb + ((size_t)(gb << 12) + npix) * 512 + ci0 + lch;
}

__launch_bounds__(256)
__global__ void conv_gemm_kernel(const bf16* __restrict__ xb, const bf16* __restrict__ WcT,
                                 float* __restrict__ part) {
  __shared__ bf16 As[2][64][32];
  const int t = threadIdx.x, lane = t & 31, wid = t >> 5;
  const int wm = wid >> 1, wn = wid & 1;
  const int row0 = blockIdx.x * 64, col0 = blockIdx.y * 128;
  const int kbeg = blockIdx.z * 2048, kend = kbeg + 2048;
  const int lr = t >> 2, lch = (t & 3) * 8;
  const int grow = row0 + lr;
  const int gb = grow >> 6, grem = grow & 63, pi = grem >> 3, pj = grem & 7;
  bf16* l0 = &As[0][lr][lch];
  bf16* l1 = &As[1][lr][lch];
  const bf16* bb = WcT + (size_t)(col0 + wn * 64) * 32768;
  v8f acc[4] = {};

  async_copy16(conv_src(xb, gb, pi, pj, kbeg, lch), l0);
  v16bf b0 = load_bfrag(bb + (size_t)0 * 16 * 32768 + kbeg, 32768, lane);
  v16bf b1 = load_bfrag(bb + (size_t)1 * 16 * 32768 + kbeg, 32768, lane);
  v16bf b2 = load_bfrag(bb + (size_t)2 * 16 * 32768 + kbeg, 32768, lane);
  v16bf b3 = load_bfrag(bb + (size_t)3 * 16 * 32768 + kbeg, 32768, lane);
  wait_async0();
  __syncthreads();

  for (int k0 = kbeg; k0 < kend; k0 += 64) {
    v16bf a = load_afrag(&As[0][wm * 16][0], 32, lane);
    const int k1 = k0 + 32;
    async_copy16(conv_src(xb, gb, pi, pj, k1, lch), l1);
    v16bf n0 = load_bfrag(bb + (size_t)0 * 16 * 32768 + k1, 32768, lane);
    v16bf n1 = load_bfrag(bb + (size_t)1 * 16 * 32768 + k1, 32768, lane);
    v16bf n2 = load_bfrag(bb + (size_t)2 * 16 * 32768 + k1, 32768, lane);
    v16bf n3 = load_bfrag(bb + (size_t)3 * 16 * 32768 + k1, 32768, lane);
    acc[0] = WMMA_BF16(a, b0, acc[0]);
    acc[1] = WMMA_BF16(a, b1, acc[1]);
    acc[2] = WMMA_BF16(a, b2, acc[2]);
    acc[3] = WMMA_BF16(a, b3, acc[3]);
    wait_async0();
    __syncthreads();

    a = load_afrag(&As[1][wm * 16][0], 32, lane);
    const int k2 = k0 + 64;
    if (k2 < kend) {
      async_copy16(conv_src(xb, gb, pi, pj, k2, lch), l0);
      b0 = load_bfrag(bb + (size_t)0 * 16 * 32768 + k2, 32768, lane);
      b1 = load_bfrag(bb + (size_t)1 * 16 * 32768 + k2, 32768, lane);
      b2 = load_bfrag(bb + (size_t)2 * 16 * 32768 + k2, 32768, lane);
      b3 = load_bfrag(bb + (size_t)3 * 16 * 32768 + k2, 32768, lane);
    }
    acc[0] = WMMA_BF16(a, n0, acc[0]);
    acc[1] = WMMA_BF16(a, n1, acc[1]);
    acc[2] = WMMA_BF16(a, n2, acc[2]);
    acc[3] = WMMA_BF16(a, n3, acc[3]);
    wait_async0();
    __syncthreads();
  }

  float* out = part + (size_t)blockIdx.z * (512 * 512);
  const int lo = lane & 15, hi = lane >> 4;
  const int rbase = row0 + wm * 16 + hi * 8;
  const int cb = col0 + wn * 64;
#pragma unroll
  for (int s = 0; s < 4; ++s)
#pragma unroll
    for (int j = 0; j < 8; ++j)
      out[(size_t)(rbase + j) * 512 + cb + s * 16 + lo] = acc[s][j];
}

// ---------------------------------------------------------------------------
// Reduce split-K partials + LayerNorm(C=512, eps=1e-3), write bf16.
// ---------------------------------------------------------------------------
__launch_bounds__(256)
__global__ void reduce_ln_kernel(const float* __restrict__ part, const float* __restrict__ gamma,
                                 const float* __restrict__ beta, bf16* __restrict__ outn) {
  const int row = blockIdx.x * 8 + (threadIdx.x >> 5);
  const int lane = threadIdx.x & 31;
  float vv[16];
  float s = 0.f, sq = 0.f;
#pragma unroll
  for (int i = 0; i < 16; ++i) {
    const int c = lane + i * 32;
    float v = 0.f;
#pragma unroll
    for (int p = 0; p < 16; ++p) v += part[(size_t)p * 262144 + (size_t)row * 512 + c];
    vv[i] = v; s += v; sq += v * v;
  }
#pragma unroll
  for (int m = 16; m >= 1; m >>= 1) {
    s  += __shfl_xor(s, m, 32);
    sq += __shfl_xor(sq, m, 32);
  }
  const float mean = s * (1.f / 512.f);
  const float var  = sq * (1.f / 512.f) - mean * mean;
  const float inv  = rsqrtf(var + 1e-3f);
#pragma unroll
  for (int i = 0; i < 16; ++i) {
    const int c = lane + i * 32;
    outn[(size_t)row * 512 + c] = (bf16)((vv[i] - mean) * inv * gamma[c] + beta[c]);
  }
}

// ---------------------------------------------------------------------------
// Attention per (b,nh): scores = q[16,64] @ k^T, softmax over M=64 in the
// WMMA C register layout, re-shape through per-wave LDS into A-fragments,
// out = attn @ v. aout may alias qb (same rows/cols, read-before-write).
// ---------------------------------------------------------------------------
__launch_bounds__(256)
__global__ void attn_kernel(const bf16* qb, const bf16* __restrict__ kvout,
                            const bf16* __restrict__ vT, bf16* aout) {
  __shared__ bf16 At[8][16][64];
  const int t = threadIdx.x, lane = t & 31, w = t >> 5;
  const int bh = blockIdx.y, b = bh >> 3, nh = bh & 7;
  const int n0 = blockIdx.x * 128 + w * 16;
  const int lo = lane & 15, hi = lane >> 4;

  v8f acc[4] = {};
  const bf16* qbase = qb + (size_t)(b * 4096 + n0) * 512 + nh * 64;
#pragma unroll
  for (int ks = 0; ks < 2; ++ks) {
    v16bf a = load_afrag(qbase + ks * 32, 512, lane);
    const bf16* kb = kvout + (size_t)(b * 64) * 1024 + nh * 64 + ks * 32;
#pragma unroll
    for (int s = 0; s < 4; ++s) {
      v16bf bb = load_bfrag(kb + (size_t)(s * 16) * 1024, 1024, lane);
      acc[s] = WMMA_BF16(a, bb, acc[s]);
    }
  }
  const float scale = 0.125f;  // 1/sqrt(64)
#pragma unroll
  for (int j = 0; j < 8; ++j) {
    float mx = -3.0e38f;
#pragma unroll
    for (int s = 0; s < 4; ++s) { acc[s][j] *= scale; mx = fmaxf(mx, acc[s][j]); }
#pragma unroll
    for (int m = 1; m < 16; m <<= 1) mx = fmaxf(mx, __shfl_xor(mx, m, 32));
    float sum = 0.f;
#pragma unroll
    for (int s = 0; s < 4; ++s) { const float e = __expf(acc[s][j] - mx); acc[s][j] = e; sum += e; }
#pragma unroll
    for (int m = 1; m < 16; m <<= 1) sum += __shfl_xor(sum, m, 32);
    const float inv = 1.0f / sum;
#pragma unroll
    for (int s = 0; s < 4; ++s) {
      acc[s][j] *= inv;
      At[w][j + 8 * hi][s * 16 + lo] = (bf16)acc[s][j];
    }
  }
  asm volatile("s_wait_dscnt 0" ::: "memory");  // LDS RAW within wave

  v8f oacc[4] = {};
  const bf16* vb = vT + (size_t)bh * 4096;  // [HD=64][M=64] row-major
#pragma unroll
  for (int ks = 0; ks < 2; ++ks) {
    v16bf a = load_afrag(&At[w][0][ks * 32], 64, lane);
#pragma unroll
    for (int s = 0; s < 4; ++s) {
      v16bf bb = load_bfrag(vb + (size_t)(s * 16) * 64 + ks * 32, 64, lane);
      oacc[s] = WMMA_BF16(a, bb, oacc[s]);
    }
  }
#pragma unroll
  for (int s = 0; s < 4; ++s)
#pragma unroll
    for (int j = 0; j < 8; ++j) {
      const int n = n0 + j + 8 * hi;
      const int c = nh * 64 + s * 16 + lo;
      aout[(size_t)(b * 4096 + n) * 512 + c] = (bf16)oacc[s][j];
    }
}

// ---------------------------------------------------------------------------
// Prep kernels: fp32 -> bf16 conversion + weight transposes
// ---------------------------------------------------------------------------
__global__ void cvt_bf16_kernel(const float* __restrict__ in, bf16* __restrict__ out, int n4) {
  const int i = blockIdx.x * 256 + threadIdx.x;
  if (i < n4) {
    f32x4 v = ((const f32x4*)in)[i];
    v4bf o;
    o.x = (bf16)v.x; o.y = (bf16)v.y; o.z = (bf16)v.z; o.w = (bf16)v.w;
    ((v4bf*)out)[i] = o;
  }
}

// out[o*K + k] = in[k*Ncols + o]
__global__ void transpose_cvt_kernel(const float* __restrict__ in, bf16* __restrict__ out,
                                     int K, int Ncols) {
  const size_t idx = (size_t)blockIdx.x * 256 + threadIdx.x;
  if (idx < (size_t)K * Ncols) {
    const int o = (int)(idx / K), k = (int)(idx % K);
    out[idx] = (bf16)in[(size_t)k * Ncols + o];
  }
}

// WqT'[(nh*64+hd)*512 + i] = Wq[i*512 + hd*8 + nh]  (head-fast channel split)
__global__ void wq_prep_kernel(const float* __restrict__ Wq, bf16* __restrict__ WqT) {
  const int idx = blockIdx.x * 256 + threadIdx.x;  // 262144
  const int op = idx >> 9, i = idx & 511;
  const int nh = op >> 6, hd = op & 63;
  WqT[idx] = (bf16)Wq[(size_t)i * 512 + hd * 8 + nh];
}

// vT[bh][hd][m] = kvout[(b*64+m)*1024 + 512 + nh*64 + hd]
__global__ void vprep_kernel(const bf16* __restrict__ kvout, bf16* __restrict__ vT) {
  const int idx = blockIdx.x * 256 + threadIdx.x;  // 64*4096
  const int bh = idx >> 12, r = idx & 4095;
  const int hd = r >> 6, m = r & 63;
  const int b = bh >> 3, nh = bh & 7;
  vT[idx] = kvout[(size_t)(b * 64 + m) * 1024 + 512 + nh * 64 + hd];
}

// ---------------------------------------------------------------------------
extern "C" void kernel_launch(void* const* d_in, const int* in_sizes, int n_in,
                              void* d_out, int out_size, void* d_ws, size_t ws_size,
                              hipStream_t stream) {
  (void)in_sizes; (void)n_in; (void)out_size; (void)ws_size;
  const float* x     = (const float*)d_in[0];
  const float* Wq    = (const float*)d_in[1];
  const float* Wkv   = (const float*)d_in[2];
  const float* convw = (const float*)d_in[3];
  const float* gamma = (const float*)d_in[4];
  const float* beta  = (const float*)d_in[5];
  const float* Wp    = (const float*)d_in[6];
  const float* bp    = (const float*)d_in[7];
  float* out = (float*)d_out;

  char* ws = (char*)d_ws;
  size_t off = 0;
  auto alloc = [&](size_t bytes) -> void* {
    void* p = ws + off;
    off += (bytes + 255) & ~(size_t)255;
    return p;
  };
  bf16* xb    = (bf16*)alloc(16777216ull * 2);   // x in bf16 [B*N, C]
  bf16* WqT   = (bf16*)alloc(262144ull * 2);     // [512,512] permuted-transposed
  bf16* WkvT  = (bf16*)alloc(524288ull * 2);     // [1024,512]
  bf16* WcT   = (bf16*)alloc(16777216ull * 2);   // [512,32768]
  bf16* WpT   = (bf16*)alloc(262144ull * 2);     // [512,512]
  bf16* qb    = (bf16*)alloc(16777216ull * 2);   // q [B*N, (nh,hd)]; reused as attn out
  float* part = (float*)alloc(16ull * 262144 * 4);  // conv split-K partials
  bf16* fmapn = (bf16*)alloc(262144ull * 2);     // LN(conv) bf16 [512,512]
  bf16* kvout = (bf16*)alloc(524288ull * 2);     // [B*64, 1024]
  bf16* vT    = (bf16*)alloc(262144ull * 2);     // [B*NH, 64(hd), 64(m)]
  bf16* aout  = qb;                              // alias: safe read-before-write per wave

  // prep: convert + transpose into WMMA-friendly bf16 layouts
  cvt_bf16_kernel<<<16384, 256, 0, stream>>>(x, xb, 4194304);
  wq_prep_kernel<<<1024, 256, 0, stream>>>(Wq, WqT);
  transpose_cvt_kernel<<<2048, 256, 0, stream>>>(Wkv, WkvT, 512, 1024);
  transpose_cvt_kernel<<<65536, 256, 0, stream>>>(convw, WcT, 32768, 512);
  transpose_cvt_kernel<<<1024, 256, 0, stream>>>(Wp, WpT, 512, 512);

  // q = x @ Wq (permuted cols -> [B,N,(nh,hd)])
  gemm_bf16_kernel<0><<<dim3(512, 4), 256, 0, stream>>>(xb, WqT, qb, nullptr, 512, 512, 512, 512);
  // conv patch-GEMM, split-K=16 (deterministic partials)
  conv_gemm_kernel<<<dim3(8, 4, 16), 256, 0, stream>>>(xb, WcT, part);
  // reduce partials + layernorm
  reduce_ln_kernel<<<64, 256, 0, stream>>>(part, gamma, beta, fmapn);
  // kv = LN(fmap) @ Wkv  (native channel layout f*512 + nh*64 + hd)
  gemm_bf16_kernel<0><<<dim3(8, 8), 256, 0, stream>>>(fmapn, WkvT, kvout, nullptr, 512, 512, 512, 1024);
  // v transpose for attn@v B-operand
  vprep_kernel<<<1024, 256, 0, stream>>>(kvout, vT);
  // attention (softmax over M=64) -> aout [B,N,(nh,hd)] bf16
  attn_kernel<<<dim3(32, 64), 256, 0, stream>>>(qb, kvout, vT, aout);
  // projection + bias -> fp32 output
  gemm_bf16_kernel<1><<<dim3(512, 4), 256, 0, stream>>>(aout, WpT, out, bp, 512, 512, 512, 512);
}
